// TransformerLayerWithRPB_22290880266352
// MI455X (gfx1250) — compile-verified, matrix operand order checked by reading, not run on hardware
//
#include <hip/hip_runtime.h>
#include <hip/hip_bf16.h>
#include <math.h>

// ---------------------------------------------------------------------------
// Problem constants (match reference)
// ---------------------------------------------------------------------------
#define BB   4
#define SS   2048
#define DD   256
#define HH   8
#define DHH  32
#define DFFF 1024

typedef __attribute__((ext_vector_type(16))) _Float16 vhalf16;
typedef __attribute__((ext_vector_type(8)))  _Float16 vhalf8;
typedef __attribute__((ext_vector_type(8)))  float    vfloat8;

// Load one 16-element f16 fragment: elements 0..7 from p[0..7], 8..15 from
// p[16..23] (CDNA5 16-bit A/B fragment per-lane pattern; caller bakes in the
// 8*(lane/16) half offset).
__device__ __forceinline__ vhalf16 load_frag16(const _Float16* p) {
    vhalf16 r;
    ((vhalf8*)&r)[0] = *(const vhalf8*)(p);
    ((vhalf8*)&r)[1] = *(const vhalf8*)(p + 16);
    return r;
}

__device__ __forceinline__ vfloat8 wmma_f16(vhalf16 a, vhalf16 b, vfloat8 c) {
    // D = A(16x32 f16) * B(32x16 f16) + C(16x16 f32)
    return __builtin_amdgcn_wmma_f32_16x16x32_f16(
        /*neg_a=*/false, a, /*neg_b=*/false, b,
        /*c_mod=*/(short)0, c, /*reuse_a=*/false, /*reuse_b=*/false);
}

// ---------------------------------------------------------------------------
// DPP16 ROW_XMASK butterflies: VALU-only 16-lane reductions (stay inside the
// lane half, matching the WMMA C/D fragment layout; no DS-unit traffic).
// ctrl = 0x160 | xor_mask
// ---------------------------------------------------------------------------
#define DPP_XMASK_F(x, mask)                                                   \
    __int_as_float(__builtin_amdgcn_update_dpp(                                \
        0, __float_as_int(x), 0x160 | (mask), 0xF, 0xF, true))

__device__ __forceinline__ float rowmax16(float x) {
    x = fmaxf(x, DPP_XMASK_F(x, 1));
    x = fmaxf(x, DPP_XMASK_F(x, 2));
    x = fmaxf(x, DPP_XMASK_F(x, 4));
    x = fmaxf(x, DPP_XMASK_F(x, 8));
    return x;
}
__device__ __forceinline__ float rowsum16(float x) {
    x += DPP_XMASK_F(x, 1);
    x += DPP_XMASK_F(x, 2);
    x += DPP_XMASK_F(x, 4);
    x += DPP_XMASK_F(x, 8);
    return x;
}

// ---------------------------------------------------------------------------
// Async global -> LDS copy, 16B per lane, tracked by ASYNCcnt
// ---------------------------------------------------------------------------
__device__ __forceinline__ void async_copy_b128(unsigned lds_off, const void* gptr) {
    asm volatile("global_load_async_to_lds_b128 %0, %1, off"
                 :: "v"(lds_off), "v"(gptr)
                 : "memory");
}

// ---------------------------------------------------------------------------
// fp32 -> f16 elementwise convert (weights, once per launch)
// ---------------------------------------------------------------------------
__global__ __launch_bounds__(256) void f2h_ker(const float* __restrict__ in,
                                               _Float16* __restrict__ out, int n) {
    int i = blockIdx.x * 256 + threadIdx.x;
    if (i < n) out[i] = (_Float16)in[i];
}

// ---------------------------------------------------------------------------
// LayerNorm over D=256, one row per block (256 threads = 8 waves), f16 out
// ---------------------------------------------------------------------------
__global__ __launch_bounds__(256) void ln_ker(const float* __restrict__ x,
                                              const float* __restrict__ w,
                                              const float* __restrict__ b,
                                              _Float16* __restrict__ out) {
    const int t = threadIdx.x;
    const size_t row = blockIdx.x;
    float v = x[row * DD + t];

    __shared__ float red1[8];
    __shared__ float red2[8];

    // wave sum (full 32 lanes): two half-reductions + one cross-half DPP is
    // possible, but a plain 5-stage xmask butterfly needs the 16-swap too, so
    // do 4 xmask stages then combine halves via permlanex-free shared path.
    float s = rowsum16(v);
    // fold the two halves of the wave through LDS: lane half 0
    // and 1 both write their partial; use shared since only 8 waves.
    __shared__ float half1[16];  // [wave][half]
    if ((t & 15) == 0) half1[((t >> 5) << 1) | ((t >> 4) & 1)] = s;
    __syncthreads();
    if (t < 8) red1[t] = half1[2 * t] + half1[2 * t + 1];
    __syncthreads();
    float tot = 0.f;
    #pragma unroll
    for (int i = 0; i < 8; ++i) tot += red1[i];
    const float mu = tot * (1.0f / DD);

    const float d = v - mu;
    float s2 = rowsum16(d * d);
    __shared__ float half2[16];
    if ((t & 15) == 0) half2[((t >> 5) << 1) | ((t >> 4) & 1)] = s2;
    __syncthreads();
    if (t < 8) red2[t] = half2[2 * t] + half2[2 * t + 1];
    __syncthreads();
    float tot2 = 0.f;
    #pragma unroll
    for (int i = 0; i < 8; ++i) tot2 += red2[i];
    const float var = tot2 * (1.0f / DD);

    const float y = d * rsqrtf(var + 1e-5f) * w[t] + b[t];
    out[row * DD + t] = (_Float16)y;
}

// ---------------------------------------------------------------------------
// WMMA GEMM: Y[M,N] = A[M,K](f16) * W[N,K]^T(f16) + bias
// optional exact GELU, optional fp32 residual add, f16 or f32 output.
// Block: 256 threads = 8 waves; block tile 64 rows x 128 cols; wave tile
// 16 rows x 64 cols. The 128x32 W tile (8KB) is staged into LDS with
// double-buffered async global->LDS b128 copies (ASYNCcnt), so the four
// row-waves share one global fetch of W; B-frags come from LDS.
// ---------------------------------------------------------------------------
template <bool GELU, bool RESID, bool OUT_F16>
__global__ __launch_bounds__(256) void gemm_ker(const _Float16* __restrict__ A,
                                                const _Float16* __restrict__ W,
                                                const float* __restrict__ bias,
                                                const float* __restrict__ resid,
                                                void* __restrict__ outp,
                                                int N, int K) {
    __shared__ __align__(16) _Float16 wt[2][128][32];

    const int t = threadIdx.x;
    const int lane = t & 31;
    const int wave = t >> 5;
    const int lr = lane & 15;   // row (A) / col (B) within 16-tile
    const int hf = lane >> 4;   // lane half -> K sub-offset
    const int row0 = blockIdx.y * 64 + (wave & 3) * 16;
    const int cb0  = blockIdx.x * 128;
    const int cw   = (wave >> 2) * 64;

    // copy-lane mapping: 256 threads x 2 chunks x 16B = 8KB tile
    const int cr = t >> 2;  // 0..63  (also +64)
    const int cc = t & 3;   // 16B chunk within a 64B row
    const int nk = K >> 5;

    auto issue = [&](int buf, int k0) {
        const _Float16* g0 = W + (size_t)(cb0 + cr) * K + k0 + cc * 8;
        const _Float16* g1 = W + (size_t)(cb0 + cr + 64) * K + k0 + cc * 8;
        unsigned l0 = (unsigned)(unsigned long long)&wt[buf][cr][cc * 8];
        unsigned l1 = (unsigned)(unsigned long long)&wt[buf][cr + 64][cc * 8];
        async_copy_b128(l0, g0);
        async_copy_b128(l1, g1);
    };

    vfloat8 acc[4] = {};
    issue(0, 0);
    for (int it = 0; it < nk; ++it) {
        const int k0 = it << 5;
        const int buf = it & 1;
        if (it + 1 < nk) {
            issue(buf ^ 1, k0 + 32);
            asm volatile("s_wait_asynccnt 0x2" ::: "memory");  // current tile done
        } else {
            asm volatile("s_wait_asynccnt 0x0" ::: "memory");
        }
        __syncthreads();

        const _Float16* pa = A + (size_t)(row0 + lr) * K + k0 + 8 * hf;
        vhalf16 af = load_frag16(pa);
        #pragma unroll
        for (int j = 0; j < 4; ++j) {
            vhalf16 bf = load_frag16(&wt[buf][cw + 16 * j + lr][8 * hf]);
            acc[j] = wmma_f16(af, bf, acc[j]);
        }
        __syncthreads();  // frag reads done before buf^1 is overwritten
    }

    #pragma unroll
    for (int j = 0; j < 4; ++j) {
        const int gcol = cb0 + cw + 16 * j + lr;
        const float bv = bias[gcol];
        #pragma unroll
        for (int r = 0; r < 8; ++r) {
            const int grow = row0 + r + 8 * hf;
            const size_t idx = (size_t)grow * N + gcol;
            float v = acc[j][r] + bv;
            if (GELU)  v = 0.5f * v * (1.0f + erff(v * 0.70710678118654752f));
            if (RESID) v += resid[idx];
            if (OUT_F16) ((_Float16*)outp)[idx] = (_Float16)v;
            else         ((float*)outp)[idx] = v;
        }
    }
}

// ---------------------------------------------------------------------------
// Split fused qkv[BS, 3D] into q[B,H,S,DH], k[B,H,S,DH], vT[B,H,DH,S]
// ---------------------------------------------------------------------------
__global__ __launch_bounds__(256) void reshape_qkv_ker(const _Float16* __restrict__ qkv,
                                                       _Float16* __restrict__ q,
                                                       _Float16* __restrict__ k,
                                                       _Float16* __restrict__ vT) {
    const size_t i = (size_t)blockIdx.x * 256 + threadIdx.x;  // over B*S*3D
    const int e = (int)(i % (3 * DD));
    const size_t bs = i / (3 * DD);
    const int s = (int)(bs % SS);
    const int b = (int)(bs / SS);
    const int which = e >> 8;        // 0=q 1=k 2=v
    const int d = e & (DD - 1);
    const int h = d >> 5;
    const int dh = d & (DHH - 1);
    const _Float16 val = qkv[i];
    const size_t bh = (size_t)b * HH + h;
    if (which == 0)      q [(bh * SS + s) * DHH + dh] = val;
    else if (which == 1) k [(bh * SS + s) * DHH + dh] = val;
    else                 vT[(bh * DHH + dh) * SS + s] = val;
}

// ---------------------------------------------------------------------------
// Flash attention with ALiBi-style bias. One wave owns 16 query rows; block =
// 8 waves = 128 rows; grid = (B*H, S/128). Single pass over keys in steps of
// 32 (2 score WMMAs + 2 PV WMMAs per step), online softmax in fp32 with
// DPP16 xmask row reductions (VALU) and a wave-private LDS P-transpose.
// ---------------------------------------------------------------------------
__global__ __launch_bounds__(256) void attn_ker(const _Float16* __restrict__ qg,
                                                const _Float16* __restrict__ kg,
                                                const _Float16* __restrict__ vTg,
                                                const float* __restrict__ slopes,
                                                _Float16* __restrict__ ctx) {
    const int bh = blockIdx.x;            // 0..B*H-1
    const int b = bh >> 3, h = bh & 7;
    const int lane = threadIdx.x & 31;
    const int wave = threadIdx.x >> 5;
    const int lr = lane & 15;
    const int hf = lane >> 4;
    const int q0 = blockIdx.y * 128 + wave * 16;

    const _Float16* qh = qg  + (size_t)bh * SS * DHH;
    const _Float16* kh = kg  + (size_t)bh * SS * DHH;
    const _Float16* vh = vTg + (size_t)bh * DHH * SS;
    const float slope = fabsf(slopes[h]);
    const float scale = 0.17677669529663687f;  // 1/sqrt(DH)

    // Q fragment: row = q0+lr, dh elements per CDNA5 A-frag pattern
    const vhalf16 qa = load_frag16(qh + (size_t)(q0 + lr) * DHH + 8 * hf);

    vfloat8 o0 = {}, o1 = {};            // dh 0..15 / 16..31 accumulators
    float m_i[8], l_i[8];
    #pragma unroll
    for (int r = 0; r < 8; ++r) { m_i[r] = -1e30f; l_i[r] = 0.0f; }

    // wave-private P transpose tile (D-layout -> A-layout round trip)
    __shared__ __align__(16) _Float16 plds[8][16][32];
    _Float16 (*pt)[32] = plds[wave];

    for (int kb = 0; kb < SS; kb += 32) {
        // ---- scores: two 16x16 tiles, contraction = DH = 32 (one WMMA each)
        vhalf16 kb0 = load_frag16(kh + (size_t)(kb + lr) * DHH + 8 * hf);
        vhalf16 kb1 = load_frag16(kh + (size_t)(kb + 16 + lr) * DHH + 8 * hf);
        vfloat8 s0 = {}, s1 = {};
        s0 = wmma_f16(qa, kb0, s0);
        s1 = wmma_f16(qa, kb1, s1);

        // ---- scale + ALiBi bias (D-frag: row = r+8*hf, col = lr)
        #pragma unroll
        for (int r = 0; r < 8; ++r) {
            const int qpos = q0 + r + 8 * hf;
            s0[r] = s0[r] * scale - slope * fabsf((float)(qpos - (kb + lr)));
            s1[r] = s1[r] * scale - slope * fabsf((float)(qpos - (kb + 16 + lr)));
        }

        // ---- online softmax: DPP16 xmask butterflies (stay inside the half)
        #pragma unroll
        for (int r = 0; r < 8; ++r) {
            const float mt = rowmax16(fmaxf(s0[r], s1[r]));
            const float mnew = fmaxf(m_i[r], mt);
            const float f = __expf(m_i[r] - mnew);
            const float p0 = __expf(s0[r] - mnew);
            const float p1 = __expf(s1[r] - mnew);
            const float ps = rowsum16(p0 + p1);
            l_i[r] = l_i[r] * f + ps;
            m_i[r] = mnew;
            o0[r] = o0[r] * f;
            o1[r] = o1[r] * f;
            pt[r + 8 * hf][lr]      = (_Float16)p0;
            pt[r + 8 * hf][lr + 16] = (_Float16)p1;
        }
        // CDNA5 split dependency counter: order LDS store -> LDS load
        asm volatile("s_wait_dscnt 0" ::: "memory");

        // ---- P as A-frag (from LDS), V^T as B-frags (contiguous over keys)
        vhalf16 pa  = load_frag16(&pt[lr][8 * hf]);
        vhalf16 vb0 = load_frag16(vh + (size_t)lr        * SS + kb + 8 * hf);
        vhalf16 vb1 = load_frag16(vh + (size_t)(lr + 16) * SS + kb + 8 * hf);
        o0 = wmma_f16(pa, vb0, o0);
        o1 = wmma_f16(pa, vb1, o1);
    }

    // ---- finalize: ctx[b, q, h*32 + dh]
    #pragma unroll
    for (int r = 0; r < 8; ++r) {
        const float inv = 1.0f / l_i[r];
        const int qpos = q0 + r + 8 * hf;
        const size_t base = ((size_t)b * SS + qpos) * DD + h * DHH;
        ctx[base + lr]      = (_Float16)(o0[r] * inv);
        ctx[base + lr + 16] = (_Float16)(o1[r] * inv);
    }
}

// ---------------------------------------------------------------------------
// Launch
// ---------------------------------------------------------------------------
extern "C" void kernel_launch(void* const* d_in, const int* in_sizes, int n_in,
                              void* d_out, int out_size, void* d_ws, size_t ws_size,
                              hipStream_t stream) {
    (void)in_sizes; (void)n_in; (void)out_size; (void)ws_size;

    const float* x          = (const float*)d_in[0];
    const float* in_proj_w  = (const float*)d_in[1];
    const float* in_proj_b  = (const float*)d_in[2];
    const float* out_w      = (const float*)d_in[3];
    const float* out_b      = (const float*)d_in[4];
    const float* ln1_w      = (const float*)d_in[5];
    const float* ln1_b      = (const float*)d_in[6];
    const float* ln2_w      = (const float*)d_in[7];
    const float* ln2_b      = (const float*)d_in[8];
    const float* ff1_w      = (const float*)d_in[9];
    const float* ff1_b      = (const float*)d_in[10];
    const float* ff2_w      = (const float*)d_in[11];
    const float* ff2_b      = (const float*)d_in[12];
    const float* slopes     = (const float*)d_in[13];
    float* out = (float*)d_out;

    const size_t BS = (size_t)BB * SS;  // 8192 rows

    // workspace carve-up (256B aligned slabs)
    char* ws = (char*)d_ws;
    size_t off = 0;
    auto take = [&](size_t bytes) -> void* {
        void* p = ws + off;
        off += (bytes + 255) & ~(size_t)255;
        return p;
    };
    _Float16* xn_h   = (_Float16*)take(BS * DD * 2);
    _Float16* qkv_h  = (_Float16*)take(BS * 3 * DD * 2);
    _Float16* q_h    = (_Float16*)take((size_t)BB * HH * SS * DHH * 2);
    _Float16* k_h    = (_Float16*)take((size_t)BB * HH * SS * DHH * 2);
    _Float16* vT_h   = (_Float16*)take((size_t)BB * HH * DHH * SS * 2);
    _Float16* ctx_h  = (_Float16*)take(BS * DD * 2);
    float*    x2     = (float*)   take(BS * DD * 4);
    _Float16* xn2_h  = (_Float16*)take(BS * DD * 2);
    _Float16* h_h    = (_Float16*)take(BS * DFFF * 2);
    _Float16* w_in_h = (_Float16*)take((size_t)3 * DD * DD * 2);
    _Float16* w_out_h= (_Float16*)take((size_t)DD * DD * 2);
    _Float16* w_f1_h = (_Float16*)take((size_t)DFFF * DD * 2);
    _Float16* w_f2_h = (_Float16*)take((size_t)DD * DFFF * 2);

    // 1) weights -> f16
    f2h_ker<<<(3 * DD * DD) / 256, 256, 0, stream>>>(in_proj_w, w_in_h, 3 * DD * DD);
    f2h_ker<<<(DD * DD) / 256, 256, 0, stream>>>(out_w, w_out_h, DD * DD);
    f2h_ker<<<(DFFF * DD) / 256, 256, 0, stream>>>(ff1_w, w_f1_h, DFFF * DD);
    f2h_ker<<<(DD * DFFF) / 256, 256, 0, stream>>>(ff2_w, w_f2_h, DD * DFFF);

    // 2) LN1
    ln_ker<<<(int)BS, 256, 0, stream>>>(x, ln1_w, ln1_b, xn_h);

    // 3) QKV projection: [8192,768] = xn * in_proj_w^T + b   (f16 out)
    gemm_ker<false, false, true><<<dim3((3 * DD) / 128, (int)(BS / 64)), 256, 0, stream>>>(
        xn_h, w_in_h, in_proj_b, nullptr, qkv_h, 3 * DD, DD);

    // 4) split heads (+ V transpose)
    reshape_qkv_ker<<<(int)((BS * 3 * DD) / 256), 256, 0, stream>>>(qkv_h, q_h, k_h, vT_h);

    // 5) flash attention
    attn_ker<<<dim3(BB * HH, SS / 128), 256, 0, stream>>>(q_h, k_h, vT_h, slopes, ctx_h);

    // 6) out projection + residual (f32): x2 = x + ctx * out_w^T + out_b
    gemm_ker<false, true, false><<<dim3(DD / 128, (int)(BS / 64)), 256, 0, stream>>>(
        ctx_h, w_out_h, out_b, x, x2, DD, DD);

    // 7) LN2
    ln_ker<<<(int)BS, 256, 0, stream>>>(x2, ln2_w, ln2_b, xn2_h);

    // 8) FF1 + exact GELU (f16 out)
    gemm_ker<true, false, true><<<dim3(DFFF / 128, (int)(BS / 64)), 256, 0, stream>>>(
        xn2_h, w_f1_h, ff1_b, nullptr, h_h, DFFF, DD);

    // 9) FF2 + residual -> final output (f32)
    gemm_ker<false, true, false><<<dim3(DD / 128, (int)(BS / 64)), 256, 0, stream>>>(
        h_h, w_f2_h, ff2_b, x2, out, DD, DFFF);
}
